// ChessRelativeAttention_78280073937250
// MI455X (gfx1250) — matrix operand order repaired and checked by reading, not compiled, not run on hardware
//
#include <hip/hip_runtime.h>

// ---------------------------------------------------------------------------
// Types
// ---------------------------------------------------------------------------
typedef __attribute__((ext_vector_type(16))) __bf16 v16bf;
typedef __attribute__((ext_vector_type(8)))  float  v8f;
typedef __attribute__((ext_vector_type(4)))  __bf16 v4bf;
typedef int gv4i __attribute__((vector_size(16)));   // matches builtin param type

__device__ __forceinline__ __bf16 f2bf(float f) { return (__bf16)f; }
__device__ __forceinline__ float  bf2f(__bf16 h) { return (float)h; }

__device__ __forceinline__ v8f wmma_bf16(v16bf a, v16bf b, v8f c) {
    return __builtin_amdgcn_wmma_f32_16x16x32_bf16(
        /*neg_a=*/false, a, /*neg_b=*/false, b,
        /*c_mod=*/(short)0, c, /*reuse_a=*/false, /*reuse_b=*/false);
}

// ---------------------------------------------------------------------------
// Async global->LDS copy (CDNA5 GLOBAL_LOAD_ASYNC_TO_LDS_B128, ASYNCcnt),
// with a synchronous fallback so the file compiles on any toolchain.
// Builtin signature (from hipcc diagnostic): param0 = global (AS1) int4*,
// param1 = LDS (AS3) int4*, then imm offset, imm cpol.
// ---------------------------------------------------------------------------
#if defined(__has_builtin)
#  if __has_builtin(__builtin_amdgcn_global_load_async_to_lds_b128)
#    define HAS_ASYNC_LDS 1
#  endif
#endif
#ifndef HAS_ASYNC_LDS
#  define HAS_ASYNC_LDS 0
#endif

__device__ __forceinline__ void async_copy16(void* lds_dst, const void* gsrc) {
#if HAS_ASYNC_LDS
    __builtin_amdgcn_global_load_async_to_lds_b128(
        (__attribute__((address_space(1))) gv4i*)gsrc,
        (__attribute__((address_space(3))) gv4i*)lds_dst,
        /*offset=*/0, /*cpol=*/0);
#else
    *(float4*)lds_dst = *(const float4*)gsrc;
#endif
}

template <int N>
__device__ __forceinline__ void wait_async() {
#if HAS_ASYNC_LDS
#  if defined(__has_builtin) && __has_builtin(__builtin_amdgcn_s_wait_asynccnt)
    __builtin_amdgcn_s_wait_asynccnt(N);
#  else
    asm volatile("s_wait_asynccnt %0" :: "n"(N) : "memory");
#  endif
#endif
}

// ---------------------------------------------------------------------------
// Problem constants
// ---------------------------------------------------------------------------
#define DM    1024      // d_model
#define NHEAD 16
#define HDIM  64
#define SEQ   64
#define BATCH 1024
#define NTOK  (BATCH * SEQ)                 // 65536
static constexpr size_t QKV_MAT_ELEMS = (size_t)NTOK * DM;   // 67,108,864 bf16 per matrix

// Workspace layout (bytes)
static constexpr size_t WT_QKV_OFF = 0;                                   // 3*1024*1024 bf16
static constexpr size_t WT_O_OFF   = WT_QKV_OFF + (size_t)3*DM*DM*2;      // 1024*1024 bf16
static constexpr size_t XB_OFF     = WT_O_OFF   + (size_t)DM*DM*2;        // NTOK*DM bf16
static constexpr size_t QKV_OFF    = XB_OFF     + QKV_MAT_ELEMS*2;        // 3 * NTOK*DM bf16
static constexpr size_t OA_OFF     = QKV_OFF    + 3*QKV_MAT_ELEMS*2;      // NTOK*DM bf16

// ---------------------------------------------------------------------------
// Fragment loaders (pitch in bf16 elements; 16B-aligned rows required)
//   A 16x32 bf16: lane holds row M=lane&15; K chunks {base..+7, base+16..+23},
//                 base = (lane&16)?8:0    (ISA 7.12.2, 16-bit A 16x32)
//   B 32x16 bf16: lane holds col N=lane&15; K chunks {base..+7, base+16..+23},
//                 base = (lane&16)?8:0    (same K order as A)
// ---------------------------------------------------------------------------
__device__ __forceinline__ v16bf frag_a(const __bf16* tile, int pitch, int k0) {
    int lane = threadIdx.x & 31;
    const __bf16* p = tile + (size_t)(lane & 15) * pitch + k0 + ((lane & 16) ? 8 : 0);
    union { v16bf v; float4 f[2]; } u;
    u.f[0] = *(const float4*)p;
    u.f[1] = *(const float4*)(p + 16);
    return u.v;
}

__device__ __forceinline__ v16bf frag_b(const __bf16* tile, int pitch, int k0) {
    int lane = threadIdx.x & 31;
    const __bf16* p = tile + (size_t)(lane & 15) * pitch + k0 + ((lane & 16) ? 8 : 0);
    union { v16bf v; float4 f[2]; } u;
    u.f[0] = *(const float4*)p;
    u.f[1] = *(const float4*)(p + 16);
    return u.v;
}

#define GPITCH 40     // GEMM LDS tile pitch: 32 K + 8 pad (80 B rows)
#define APITCH 72     // attention LDS tile pitch: 64 + 8 pad (144 B rows)

// ---------------------------------------------------------------------------
// Kernel: fp32 [1024][1024] -> bf16 transposed [col][row]
// ---------------------------------------------------------------------------
__global__ __launch_bounds__(256) void k_transpose_convert(
    const float* __restrict__ src, __bf16* __restrict__ dst)
{
    __shared__ float tile[32][33];
    int tx = threadIdx.x, ty = threadIdx.y;
    int c0 = blockIdx.x * 32;   // source col block
    int r0 = blockIdx.y * 32;   // source row block
#pragma unroll
    for (int i = ty; i < 32; i += 8)
        tile[i][tx] = src[(size_t)(r0 + i) * DM + c0 + tx];
    __syncthreads();
#pragma unroll
    for (int i = ty; i < 32; i += 8)
        dst[(size_t)(c0 + i) * DM + r0 + tx] = f2bf(tile[tx][i]);
}

// ---------------------------------------------------------------------------
// Kernel: fp32 -> bf16 elementwise (4 per thread)
// ---------------------------------------------------------------------------
__global__ __launch_bounds__(256) void k_convert_x(
    const float* __restrict__ src, __bf16* __restrict__ dst)
{
    size_t i = ((size_t)blockIdx.x * 256 + threadIdx.x) * 4;
    float4 f = *(const float4*)(src + i);
    v4bf o;
    o[0] = f2bf(f.x); o[1] = f2bf(f.y); o[2] = f2bf(f.z); o[3] = f2bf(f.w);
    *(v4bf*)(dst + i) = o;
}

// ---------------------------------------------------------------------------
// GEMM tile staging: 128x32 bf16 tiles of A and Bt -> LDS (async if possible)
// 512 chunks of 8 bf16 per tile; 2 chunks A + 2 chunks B per thread.
// ---------------------------------------------------------------------------
__device__ __forceinline__ void stage_gemm_tile(
    __bf16 (*dA)[GPITCH], __bf16 (*dB)[GPITCH],
    const __bf16* __restrict__ A, const __bf16* __restrict__ Bt,
    int M0, int N0, int K0, int tid)
{
#pragma unroll
    for (int i = 0; i < 2; ++i) {
        int c  = tid + i * 256;
        int r  = c >> 2;
        int ko = (c & 3) * 8;
        async_copy16(&dA[r][ko], A  + (size_t)(M0 + r) * DM + K0 + ko);
        async_copy16(&dB[r][ko], Bt + (size_t)(N0 + r) * DM + K0 + ko);
    }
}

// ---------------------------------------------------------------------------
// Kernel: fused QKV projection GEMM (double-buffered async pipeline)
//   C[t][o] = sum_k Xb[t][k] * Wt[o][k] + bias,  o in [0,3072)
//   Block tile 128x128, K-step 32, 8 waves in 2(M)x4(N) layout,
//   each wave owns 4x2 = 8 wmma accumulators (64x32 region).
// ---------------------------------------------------------------------------
__global__ __launch_bounds__(256) void k_qkv_gemm(
    const __bf16* __restrict__ A, const __bf16* __restrict__ Bt,
    const float* __restrict__ bq, const float* __restrict__ bk,
    const float* __restrict__ bv, __bf16* __restrict__ qkv)
{
    __shared__ __bf16 As[2][128][GPITCH];
    __shared__ __bf16 Bs[2][128][GPITCH];

    int tid  = threadIdx.x;
    int lane = tid & 31;
    int wave = tid >> 5;
    int wm = wave & 1;        // 0..1 -> 64-row half
    int wn = wave >> 1;       // 0..3 -> 32-col strip
    int M0 = blockIdx.y * 128;
    int N0 = blockIdx.x * 128;

    v8f zero = {0.f,0.f,0.f,0.f,0.f,0.f,0.f,0.f};
    v8f acc[4][2];
#pragma unroll
    for (int i = 0; i < 4; ++i)
#pragma unroll
        for (int j = 0; j < 2; ++j) acc[i][j] = zero;

    stage_gemm_tile(As[0], Bs[0], A, Bt, M0, N0, 0, tid);

    for (int kt = 0; kt < 32; ++kt) {
        int buf = kt & 1;
        if (kt < 31) {
            stage_gemm_tile(As[buf ^ 1], Bs[buf ^ 1], A, Bt, M0, N0, (kt + 1) * 32, tid);
            wait_async<4>();     // drain everything except the newest tile
        } else {
            wait_async<0>();
        }
        __syncthreads();

        v16bf af[4], bfr[2];
#pragma unroll
        for (int i = 0; i < 4; ++i) af[i]  = frag_a(&As[buf][wm * 64 + i * 16][0], GPITCH, 0);
#pragma unroll
        for (int j = 0; j < 2; ++j) bfr[j] = frag_b(&Bs[buf][wn * 32 + j * 16][0], GPITCH, 0);
#pragma unroll
        for (int i = 0; i < 4; ++i)
#pragma unroll
            for (int j = 0; j < 2; ++j)
                acc[i][j] = wmma_bf16(af[i], bfr[j], acc[i][j]);
        __syncthreads();
    }

    // Epilogue: bias + scatter to [m][(b*16+h)*64+s][d] bf16
    int m = N0 >> 10;                                  // which of Q/K/V
    const float* bias = (m == 0) ? bq : (m == 1) ? bk : bv;
    __bf16* qbase = qkv + (size_t)m * QKV_MAT_ELEMS;
#pragma unroll
    for (int i = 0; i < 4; ++i) {
        int row0 = M0 + wm * 64 + i * 16 + ((lane & 16) ? 8 : 0);
#pragma unroll
        for (int j = 0; j < 2; ++j) {
            int col = N0 + wn * 32 + j * 16 + (lane & 15);
            int rem = col & (DM - 1);
            int h = rem >> 6, d = rem & 63;
            float bval = bias[rem];
#pragma unroll
            for (int r = 0; r < 8; ++r) {
                int t = row0 + r;
                int b = t >> 6, s = t & 63;
                qbase[(((size_t)(b * NHEAD + h) * SEQ + s) << 6) + d] =
                    f2bf(acc[i][j][r] + bval);
            }
        }
    }
}

// ---------------------------------------------------------------------------
// Kernel: attention core, one block (256 thr, 8 waves) per (b,h)
// ---------------------------------------------------------------------------
__global__ __launch_bounds__(256) void k_attention(
    const __bf16* __restrict__ qkv, const float* __restrict__ Eh,
    const float* __restrict__ Ew, __bf16* __restrict__ oattn)
{
    __shared__ __bf16 qs [64][APITCH];
    __shared__ __bf16 ksm[64][APITCH];
    __shared__ __bf16 vts[64][APITCH];   // V transposed: vts[d][k]
    __shared__ __bf16 ats[64][APITCH];   // attn probs bf16
    __shared__ __bf16 ehs[16][APITCH];   // Eh rows (row 15 zero)
    __shared__ __bf16 ews[16][APITCH];
    __shared__ __bf16 p1 [64][16];
    __shared__ __bf16 p2 [64][16];
    __shared__ float  sc [64][66];       // scores fp32

    int tid  = threadIdx.x;
    int lane = tid & 31;
    int wave = tid >> 5;
    int bh = blockIdx.x;

    const __bf16* gq = qkv + (size_t)bh * (SEQ * HDIM);
    const __bf16* gk = gq + QKV_MAT_ELEMS;
    const __bf16* gv = gk + QKV_MAT_ELEMS;

    // --- stage Q, K via async 16B chunks; V transposed; Eh/Ew converted -----
#pragma unroll
    for (int i = 0; i < 2; ++i) {
        int c = tid + i * 256;            // 512 chunks of 8 bf16
        int r = c >> 3, o = (c & 7) * 8;
        async_copy16(&qs[r][o],  gq + r * 64 + o);
        async_copy16(&ksm[r][o], gk + r * 64 + o);
    }
#pragma unroll
    for (int i = 0; i < 16; ++i) {
        int c = tid + i * 256;            // 4096 scalars
        int s = c >> 6, d = c & 63;
        vts[d][s] = gv[c];
    }
    for (int c = tid; c < 960; c += 256) {
        int r = c >> 6, d = c & 63;
        ehs[r][d] = f2bf(Eh[c]);
        ews[r][d] = f2bf(Ew[c]);
    }
    if (tid < 64) { ehs[15][tid] = f2bf(0.f); ews[15][tid] = f2bf(0.f); }
    wait_async<0>();
    __syncthreads();

    v8f zero = {0.f,0.f,0.f,0.f,0.f,0.f,0.f,0.f};

    // --- content scores (Q K^T) + P1/P2 (Q Eh^T / Q Ew^T) -------------------
    {
        int mtile = wave & 3;
        int nb    = (wave >> 2) * 2;
        const __bf16* etile = (wave & 4) ? &ews[0][0] : &ehs[0][0];
        v8f cacc[2] = {zero, zero};
        v8f pacc = zero;
#pragma unroll
        for (int ks = 0; ks < 2; ++ks) {
            int k0 = ks * 32;
            v16bf a = frag_a(&qs[mtile * 16][0], APITCH, k0);
#pragma unroll
            for (int j = 0; j < 2; ++j) {
                v16bf bb = frag_b(&ksm[(nb + j) * 16][0], APITCH, k0);
                cacc[j] = wmma_bf16(a, bb, cacc[j]);
            }
            v16bf be = frag_b(etile, APITCH, k0);
            pacc = wmma_bf16(a, be, pacc);
        }
        int q0 = mtile * 16 + ((lane & 16) ? 8 : 0);
#pragma unroll
        for (int j = 0; j < 2; ++j) {
            int kc = (nb + j) * 16 + (lane & 15);
#pragma unroll
            for (int r = 0; r < 8; ++r) sc[q0 + r][kc] = cacc[j][r];
        }
        __bf16* pdst = (wave & 4) ? &p2[0][0] : &p1[0][0];
        int col = lane & 15;
#pragma unroll
        for (int r = 0; r < 8; ++r) pdst[(q0 + r) * 16 + col] = f2bf(pacc[r]);
    }
    __syncthreads();

    // --- add gathered positional term, scale by 1/sqrt(64) ------------------
#pragma unroll
    for (int i = 0; i < 16; ++i) {
        int c = tid + i * 256;
        int q = c >> 6, k = c & 63;
        int rel = k - q;
        int rr = rel >> 3;  rr = (rr < -7) ? -7 : rr;  rr += 7;
        int rf = (rel & 7) + 7;
        sc[q][k] = (sc[q][k] + bf2f(p1[q][rr]) + bf2f(p2[q][rf])) * 0.125f;
    }
    __syncthreads();

    // --- softmax: each wave owns 8 rows; 2 elems per lane -------------------
#pragma unroll
    for (int r = 0; r < 8; ++r) {
        int q = wave * 8 + r;
        float a0 = sc[q][lane], a1 = sc[q][lane + 32];
        float mx = fmaxf(a0, a1);
#pragma unroll
        for (int off = 16; off; off >>= 1) mx = fmaxf(mx, __shfl_xor(mx, off, 32));
        float e0 = __expf(a0 - mx), e1 = __expf(a1 - mx);
        float sm = e0 + e1;
#pragma unroll
        for (int off = 16; off; off >>= 1) sm += __shfl_xor(sm, off, 32);
        float inv = 1.0f / sm;
        ats[q][lane]      = f2bf(e0 * inv);
        ats[q][lane + 32] = f2bf(e1 * inv);
    }
    __syncthreads();

    // --- O = attn @ V  (B columns come from vts rows) -----------------------
    {
        int mtile = wave & 3;
        int nb    = (wave >> 2) * 2;
        v8f oacc[2] = {zero, zero};
#pragma unroll
        for (int ks = 0; ks < 2; ++ks) {
            int k0 = ks * 32;
            v16bf a = frag_a(&ats[mtile * 16][0], APITCH, k0);
#pragma unroll
            for (int j = 0; j < 2; ++j) {
                v16bf bb = frag_b(&vts[(nb + j) * 16][0], APITCH, k0);
                oacc[j] = wmma_bf16(a, bb, oacc[j]);
            }
        }
        int b = bh >> 4, h = bh & 15;
        int q0 = mtile * 16 + ((lane & 16) ? 8 : 0);
#pragma unroll
        for (int j = 0; j < 2; ++j) {
            int d = (nb + j) * 16 + (lane & 15);
#pragma unroll
            for (int r = 0; r < 8; ++r) {
                int q = q0 + r;
                oattn[(size_t)(b * SEQ + q) * DM + h * HDIM + d] = f2bf(oacc[j][r]);
            }
        }
    }
}

// ---------------------------------------------------------------------------
// Kernel: output projection GEMM, fp32 result + bias (same pipeline)
// ---------------------------------------------------------------------------
__global__ __launch_bounds__(256) void k_out_gemm(
    const __bf16* __restrict__ A, const __bf16* __restrict__ Bt,
    const float* __restrict__ bo, float* __restrict__ out)
{
    __shared__ __bf16 As[2][128][GPITCH];
    __shared__ __bf16 Bs[2][128][GPITCH];

    int tid  = threadIdx.x;
    int lane = tid & 31;
    int wave = tid >> 5;
    int wm = wave & 1;
    int wn = wave >> 1;
    int M0 = blockIdx.y * 128;
    int N0 = blockIdx.x * 128;

    v8f zero = {0.f,0.f,0.f,0.f,0.f,0.f,0.f,0.f};
    v8f acc[4][2];
#pragma unroll
    for (int i = 0; i < 4; ++i)
#pragma unroll
        for (int j = 0; j < 2; ++j) acc[i][j] = zero;

    stage_gemm_tile(As[0], Bs[0], A, Bt, M0, N0, 0, tid);

    for (int kt = 0; kt < 32; ++kt) {
        int buf = kt & 1;
        if (kt < 31) {
            stage_gemm_tile(As[buf ^ 1], Bs[buf ^ 1], A, Bt, M0, N0, (kt + 1) * 32, tid);
            wait_async<4>();
        } else {
            wait_async<0>();
        }
        __syncthreads();

        v16bf af[4], bfr[2];
#pragma unroll
        for (int i = 0; i < 4; ++i) af[i]  = frag_a(&As[buf][wm * 64 + i * 16][0], GPITCH, 0);
#pragma unroll
        for (int j = 0; j < 2; ++j) bfr[j] = frag_b(&Bs[buf][wn * 32 + j * 16][0], GPITCH, 0);
#pragma unroll
        for (int i = 0; i < 4; ++i)
#pragma unroll
            for (int j = 0; j < 2; ++j)
                acc[i][j] = wmma_bf16(af[i], bfr[j], acc[i][j]);
        __syncthreads();
    }

#pragma unroll
    for (int i = 0; i < 4; ++i) {
        int row0 = M0 + wm * 64 + i * 16 + ((lane & 16) ? 8 : 0);
#pragma unroll
        for (int j = 0; j < 2; ++j) {
            int col = N0 + wn * 32 + j * 16 + (lane & 15);
            float bval = bo[col];
#pragma unroll
            for (int r = 0; r < 8; ++r)
                out[(size_t)(row0 + r) * DM + col] = acc[i][j][r] + bval;
        }
    }
}

// ---------------------------------------------------------------------------
// Host launcher
// ---------------------------------------------------------------------------
extern "C" void kernel_launch(void* const* d_in, const int* in_sizes, int n_in,
                              void* d_out, int out_size, void* d_ws, size_t ws_size,
                              hipStream_t stream) {
    const float* X  = (const float*)d_in[0];
    const float* Wq = (const float*)d_in[1];
    const float* bq = (const float*)d_in[2];
    const float* Wk = (const float*)d_in[3];
    const float* bk = (const float*)d_in[4];
    const float* Wv = (const float*)d_in[5];
    const float* bv = (const float*)d_in[6];
    const float* Wo = (const float*)d_in[7];
    const float* bo = (const float*)d_in[8];
    const float* Eh = (const float*)d_in[9];
    const float* Ew = (const float*)d_in[10];

    char* ws = (char*)d_ws;
    __bf16* wt_qkv = (__bf16*)(ws + WT_QKV_OFF);
    __bf16* wt_o   = (__bf16*)(ws + WT_O_OFF);
    __bf16* xb     = (__bf16*)(ws + XB_OFF);
    __bf16* qkv    = (__bf16*)(ws + QKV_OFF);
    __bf16* oattn  = (__bf16*)(ws + OA_OFF);
    float*  out    = (float*)d_out;

    dim3 tb(32, 8);
    dim3 tg(32, 32);
    k_transpose_convert<<<tg, tb, 0, stream>>>(Wq, wt_qkv);
    k_transpose_convert<<<tg, tb, 0, stream>>>(Wk, wt_qkv + (size_t)DM * DM);
    k_transpose_convert<<<tg, tb, 0, stream>>>(Wv, wt_qkv + (size_t)2 * DM * DM);
    k_transpose_convert<<<tg, tb, 0, stream>>>(Wo, wt_o);

    k_convert_x<<<(NTOK * DM) / (256 * 4), 256, 0, stream>>>(X, xb);

    k_qkv_gemm<<<dim3(3 * DM / 128, NTOK / 128), 256, 0, stream>>>(
        xb, wt_qkv, bq, bk, bv, qkv);

    k_attention<<<BATCH * NHEAD, 256, 0, stream>>>(qkv, Eh, Ew, oattn);

    k_out_gemm<<<dim3(DM / 128, NTOK / 128), 256, 0, stream>>>(
        oattn, wt_o, bo, out);
}
